// RNN_29317446762551
// MI455X (gfx1250) — matrix operand order repaired and checked by reading, not compiled, not run on hardware
//
#include <hip/hip_runtime.h>
#include <hip/hip_bf16.h>
#include <math.h>

// Problem constants (match reference)
#define BATCH  64
#define LENGTH 512
#define DIM    512
#define LATENT 1024
#define TSTEPS (LENGTH - 1)   // 511 recurrence steps
#define NWG_REC 32            // persistent workgroups for the recurrence

typedef __bf16 bf16;
typedef bf16  v16bf __attribute__((ext_vector_type(16)));
typedef bf16  v8bf  __attribute__((ext_vector_type(8)));
typedef float v8f   __attribute__((ext_vector_type(8)));

// ---- workspace layout (bytes) ----
#define OFF_WHT   0ull                                        // WhT  bf16 [1024][1024]  (B[k][n] = Wh[n][k])
#define OFF_W1T   (OFF_WHT + (size_t)LATENT * LATENT * 2)     // W1T  bf16 [512][1024]   (B[k][n] = W1[n][k])
#define OFF_H0    (OFF_W1T + (size_t)DIM * LATENT * 2)        // h ping  bf16 [64][1024]
#define OFF_H1    (OFF_H0 + (size_t)BATCH * LATENT * 2)       // h pong  bf16 [64][1024]
#define OFF_BAR   (OFF_H1 + (size_t)BATCH * LATENT * 2)       // grid barrier counter
#define OFF_XPROJ (OFF_BAR + 256)                             // xproj f32 [511][64][1024]

// ---------------------------------------------------------------------------
// Fragment loaders following CDNA5 WMMA VGPR layouts (wave32).
// A (16x32 bf16, row-major src):
//   lane<16 : M=lane,    elems 0..7 = K k0..k0+7,    elems 8..15 = K k0+16..k0+23
//   lane>=16: M=lane-16, elems 0..7 = K k0+8..k0+15, elems 8..15 = K k0+24..k0+31
__device__ __forceinline__ v16bf load_a_tile(const bf16* __restrict__ rowPtr /* = &A[m0 + (lane&15)][k0] */,
                                             int lane) {
  const bf16* p = rowPtr + ((lane >> 4) << 3);
  v8bf c0 = *(const v8bf*)p;
  v8bf c1 = *(const v8bf*)(p + 16);
  v16bf a;
#pragma unroll
  for (int i = 0; i < 8; ++i) { a[i] = c0[i]; a[i + 8] = c1[i]; }
  return a;
}

// B (32x16 bf16): lane = K row k0+lane, 16 contiguous N values per lane.
__device__ __forceinline__ v16bf load_b_tile_g(const bf16* __restrict__ base /* = &B[k0][n0] */,
                                               int ldn, int lane) {
  const bf16* p = base + (size_t)lane * ldn;
  v8bf c0 = *(const v8bf*)p;
  v8bf c1 = *(const v8bf*)(p + 8);
  v16bf b;
#pragma unroll
  for (int i = 0; i < 8; ++i) { b[i] = c0[i]; b[i + 8] = c1[i]; }
  return b;
}

// B from LDS (row stride fixed at 32 bf16): N col = lane&15, elems in the same k order as A.
__device__ __forceinline__ v16bf load_b_tile_lds(const bf16* base /* = &sB[k0*32 + nsub] */,
                                                 int lane) {
  const bf16* p = base + ((lane >> 4) << 3) * 32 + (lane & 15);
  v16bf b;
#pragma unroll
  for (int i = 0; i < 8; ++i) { b[i] = p[i * 32]; b[i + 8] = p[(i + 16) * 32]; }
  return b;
}

__device__ __forceinline__ void pack8(v16bf& a, int off, float4 lo, float4 hi) {
  a[off + 0] = (bf16)lo.x; a[off + 1] = (bf16)lo.y; a[off + 2] = (bf16)lo.z; a[off + 3] = (bf16)lo.w;
  a[off + 4] = (bf16)hi.x; a[off + 5] = (bf16)hi.y; a[off + 6] = (bf16)hi.z; a[off + 7] = (bf16)hi.w;
}

// ---------------------------------------------------------------------------
// Kernel 1: convert + transpose weights to bf16, zero the grid barrier.
__global__ void __launch_bounds__(256) rnn_prep(const float* __restrict__ W1,
                                                const float* __restrict__ Wh,
                                                bf16* __restrict__ whT,
                                                bf16* __restrict__ w1T,
                                                unsigned* __restrict__ bar) {
  unsigned idx = blockIdx.x * blockDim.x + threadIdx.x;
  if (idx == 0)
    __hip_atomic_store(bar, 0u, __ATOMIC_RELAXED, __HIP_MEMORY_SCOPE_AGENT);
  if (idx < (unsigned)(LATENT * LATENT)) {
    unsigned k = idx >> 10, n = idx & (LATENT - 1);
    whT[idx] = (bf16)Wh[(size_t)n * LATENT + k];
  } else {
    unsigned j = idx - (unsigned)(LATENT * LATENT);
    if (j < (unsigned)(DIM * LATENT)) {
      unsigned k = j >> 10, n = j & (LATENT - 1);
      w1T[j] = (bf16)W1[(size_t)n * DIM + k];
    }
  }
}

// ---------------------------------------------------------------------------
// Kernel 2: xproj[t][b][l] = sum_d x[b][t][d] * W1[l][d] + b1[l]   (t < 511)
// Grid: (16 n-tiles of 64, 511 timesteps), 128 threads = 4 waves.
__global__ void __launch_bounds__(128) rnn_xproj(const float* __restrict__ x,
                                                 const bf16* __restrict__ w1T,
                                                 const float* __restrict__ b1,
                                                 float* __restrict__ xproj) {
  const int t = blockIdx.y;
  const int n0 = blockIdx.x * 64;
  const int lane = threadIdx.x & 31;
  const int m0 = (threadIdx.x >> 5) * 16;

  v8f acc[4] = {};
  const int bRow = m0 + (lane & 15);
  const float* aRow = x + ((size_t)bRow * LENGTH + t) * DIM;

  for (int k0 = 0; k0 < DIM; k0 += 32) {
    const float* ap = aRow + k0 + ((lane >> 4) << 3);
    float4 f0 = *(const float4*)ap;
    float4 f1 = *(const float4*)(ap + 4);
    float4 f2 = *(const float4*)(ap + 16);
    float4 f3 = *(const float4*)(ap + 20);
    v16bf a;
    pack8(a, 0, f0, f1);
    pack8(a, 8, f2, f3);

    const bf16* bBase = w1T + (size_t)(k0 + ((lane >> 4) << 3)) * LATENT + n0 + (lane & 15);
#pragma unroll
    for (int j = 0; j < 4; ++j) {
      v16bf b;
#pragma unroll
      for (int i = 0; i < 8; ++i) { b[i] = bBase[(size_t)i * LATENT + j * 16]; b[i + 8] = bBase[(size_t)(i + 16) * LATENT + j * 16]; }
      acc[j] = __builtin_amdgcn_wmma_f32_16x16x32_bf16(false, a, false, b,
                                                       (short)0, acc[j], false, false);
    }
  }

  const int mBase = m0 + ((lane >> 4) << 3);
#pragma unroll
  for (int j = 0; j < 4; ++j) {
    const int n = n0 + j * 16 + (lane & 15);
    const float bias = b1[n];
    float* orow = xproj + (size_t)t * BATCH * LATENT + n;
#pragma unroll
    for (int r = 0; r < 8; ++r)
      orow[(size_t)(mBase + r) * LATENT] = acc[j][r] + bias;
  }
}

// ---------------------------------------------------------------------------
// Kernel 3: persistent recurrence. 32 WGs x 256 threads = 256 waves; each WG
// owns a 64x32 column strip of h and caches its Wh^T slice (1024x32 bf16 =
// 64 KB) in LDS once; each wave computes one 16x16 tile per step (32 wmma).
// Steps separated by a device-scope atomic grid barrier.
__global__ void __launch_bounds__(256, 1) rnn_recurrence(const float* __restrict__ xproj,
                                                         const float* __restrict__ bh,
                                                         const bf16* __restrict__ whT,
                                                         bf16* __restrict__ h0buf,
                                                         bf16* __restrict__ h1buf,
                                                         unsigned* __restrict__ bar,
                                                         float* __restrict__ out) {
  __shared__ bf16 sB[LATENT * 32];     // WhT[0:1024, nblk:nblk+32], 64 KB

  const int lane = threadIdx.x & 31;
  const int wv = threadIdx.x >> 5;     // 0..7
  const int nblk = blockIdx.x * 32;    // this WG's 32 output columns
  const int m0 = (wv & 3) * 16;        // wave's 16 output rows
  const int nsub = (wv >> 2) * 16;     // wave's 16 cols within the strip
  const int n = nblk + nsub + (lane & 15);  // this lane's output column
  const int mBase = m0 + ((lane >> 4) << 3);

  // ---- stage B slice into LDS (once; reused by all 511 steps) ----
  {
    uint4* dst = (uint4*)sB;
    // 1024 rows x 32 bf16 = 4096 uint4; 16 per thread
    for (int i = threadIdx.x; i < LATENT * 4; i += 256) {
      const int row = i >> 2, c = (i & 3) * 8;
      dst[i] = *(const uint4*)(whT + (size_t)row * LATENT + nblk + c);
    }
  }

  // ---- h(0) = 0 : 8192 threads x 16B covers 64*1024 bf16 exactly ----
  {
    uint4 z = make_uint4(0u, 0u, 0u, 0u);
    ((uint4*)h0buf)[blockIdx.x * blockDim.x + threadIdx.x] = z;
  }
  __syncthreads();

  unsigned barPhase = 0;
  auto gsync = [&]() {
    ++barPhase;
    __syncthreads();
    if (threadIdx.x == 0) {
      __threadfence();
      __hip_atomic_fetch_add(bar, 1u, __ATOMIC_ACQ_REL, __HIP_MEMORY_SCOPE_AGENT);
      const unsigned tgt = NWG_REC * barPhase;
      while (__hip_atomic_load(bar, __ATOMIC_ACQUIRE, __HIP_MEMORY_SCOPE_AGENT) < tgt)
        __builtin_amdgcn_s_sleep(1);
      __threadfence();
    }
    __syncthreads();
  };

  bf16* bufs[2] = {h0buf, h1buf};
  const float biasN = bh[n];
  gsync();  // h(0) visible everywhere

  for (int t = 0; t < TSTEPS; ++t) {
    const bf16* cur = bufs[t & 1];
    bf16* nxt = bufs[(t + 1) & 1];
    const float* xp = xproj + (size_t)t * BATCH * LATENT;

    // hide epilogue operand latency behind the K loop
    __builtin_prefetch(&xp[(size_t)mBase * LATENT + n], 0, 0);
    __builtin_prefetch(&xp[(size_t)(mBase + 4) * LATENT + n], 0, 0);

    v8f acc = {};
    const bf16* aRow = cur + (size_t)(m0 + (lane & 15)) * LATENT;
    for (int k0 = 0; k0 < LATENT; k0 += 32) {
      v16bf a = load_a_tile(aRow + k0, lane);
      v16bf b = load_b_tile_lds(sB + k0 * 32 + nsub, lane);
      acc = __builtin_amdgcn_wmma_f32_16x16x32_bf16(false, a, false, b,
                                                    (short)0, acc, false, false);
    }

#pragma unroll
    for (int r = 0; r < 8; ++r) {
      const int m = mBase + r;
      const float v = acc[r] + xp[(size_t)m * LATENT + n] + biasN;
      nxt[(size_t)m * LATENT + n] = (bf16)tanhf(v);
    }
    gsync();  // h(t+1) complete before anyone reads it
  }

  // ---- final: y = h_final @ Wh^T + bh  (fp32 out, no tanh) ----
  {
    const bf16* cur = bufs[TSTEPS & 1];
    v8f acc = {};
    const bf16* aRow = cur + (size_t)(m0 + (lane & 15)) * LATENT;
    for (int k0 = 0; k0 < LATENT; k0 += 32) {
      v16bf a = load_a_tile(aRow + k0, lane);
      v16bf b = load_b_tile_lds(sB + k0 * 32 + nsub, lane);
      acc = __builtin_amdgcn_wmma_f32_16x16x32_bf16(false, a, false, b,
                                                    (short)0, acc, false, false);
    }
#pragma unroll
    for (int r = 0; r < 8; ++r)
      out[(size_t)(mBase + r) * LATENT + n] = acc[r] + biasN;
  }
}

// ---------------------------------------------------------------------------
extern "C" void kernel_launch(void* const* d_in, const int* in_sizes, int n_in,
                              void* d_out, int out_size, void* d_ws, size_t ws_size,
                              hipStream_t stream) {
  const float* x  = (const float*)d_in[0];
  const float* W1 = (const float*)d_in[1];
  const float* b1 = (const float*)d_in[2];
  const float* Wh = (const float*)d_in[3];
  const float* bh = (const float*)d_in[4];

  char* ws = (char*)d_ws;
  bf16* whT = (bf16*)(ws + OFF_WHT);
  bf16* w1T = (bf16*)(ws + OFF_W1T);
  bf16* h0  = (bf16*)(ws + OFF_H0);
  bf16* h1  = (bf16*)(ws + OFF_H1);
  unsigned* bar = (unsigned*)(ws + OFF_BAR);
  float* xproj = (float*)(ws + OFF_XPROJ);

  const int prepElems = LATENT * LATENT + DIM * LATENT;
  rnn_prep<<<dim3((prepElems + 255) / 256), 256, 0, stream>>>(W1, Wh, whT, w1T, bar);
  rnn_xproj<<<dim3(LATENT / 64, TSTEPS), 128, 0, stream>>>(x, w1T, b1, xproj);
  rnn_recurrence<<<NWG_REC, 256, 0, stream>>>(xproj, bh, whT, h0, h1, bar,
                                              (float*)d_out);
}
